// Linear_67070209294813
// MI455X (gfx1250) — compile-verified, matrix operand order checked by reading, not run on hardware
//
#include <hip/hip_runtime.h>
#include <stdint.h>

// Problem constants (from reference)
#define M_TOT  8192     // B*S
#define K_DIM  4096     // D_IN
#define N_DIM  4096     // D_OUT
#define R_     16
#define E_     4
#define ER_    64       // E*R  (combined lora K extension)
#define SPLIT_ 32
#define S_     2048
#define SCALING_ 2.0f   // 32/16

typedef __attribute__((ext_vector_type(16))) __bf16    v16bf;
typedef __attribute__((ext_vector_type(8)))  __bf16    v8bf;
typedef __attribute__((ext_vector_type(8)))  float     v8f;
typedef __attribute__((ext_vector_type(4)))  uint32_t  v4u;
typedef __attribute__((ext_vector_type(8)))  int       v8i;
typedef __attribute__((ext_vector_type(4)))  int       v4i;

union FragAB { v16bf v; struct { v8bf lo, hi; } p; };

#define AS1 __attribute__((address_space(1)))
#define AS3 __attribute__((address_space(3)))

// Data-mover path selection (compile-guarded so the file always builds)
#if defined(__gfx1250__) && __has_builtin(__builtin_amdgcn_tensor_load_to_lds) && __has_builtin(__builtin_amdgcn_s_wait_tensorcnt)
  #define GEMM_TDM 1
#elif defined(__gfx1250__) && __has_builtin(__builtin_amdgcn_global_load_async_to_lds_b128) && __has_builtin(__builtin_amdgcn_s_wait_asynccnt)
  #define GEMM_ASYNC 1
#endif

__device__ __forceinline__ uint32_t f32_bits(float f){ union{float f;uint32_t u;}c; c.f=f; return c.u; }
__device__ __forceinline__ float bits_f32(uint32_t u){ union{float f;uint32_t u;}c; c.u=u; return c.f; }
// round-to-nearest-even fp32 -> bf16
__device__ __forceinline__ uint16_t f2bf(float f){
  uint32_t u = f32_bits(f);
  return (uint16_t)((u + 0x7FFFu + ((u >> 16) & 1u)) >> 16);
}
__device__ __forceinline__ float bf2f(uint16_t b){ return bits_f32(((uint32_t)b) << 16); }

// generic LDS pointer -> 32-bit LDS offset (flat LDS aperture keeps offset in addr[31:0])
__device__ __forceinline__ uint32_t lds_addr32(const void* p){
  return (uint32_t)(uintptr_t)p;
}

// ---------------------------------------------------------------------------
// 1) fp32 -> bf16 hi/lo split (lo may be null -> plain bf16 path)
// ---------------------------------------------------------------------------
__global__ void split_bf16_kernel(const float* __restrict__ src,
                                  uint16_t* __restrict__ hi,
                                  uint16_t* __restrict__ lo, int n4){
  int i = blockIdx.x * blockDim.x + threadIdx.x;
  if (i >= n4) return;
  float4 v = ((const float4*)src)[i];
  float fa[4] = {v.x, v.y, v.z, v.w};
  uint16_t hb[4], lb[4];
#pragma unroll
  for (int j = 0; j < 4; ++j){
    hb[j] = f2bf(fa[j]);
    lb[j] = f2bf(fa[j] - bf2f(hb[j]));
  }
  ushort4 h; h.x=hb[0]; h.y=hb[1]; h.z=hb[2]; h.w=hb[3];
  ((ushort4*)hi)[i] = h;
  if (lo){
    ushort4 l; l.x=lb[0]; l.y=lb[1]; l.z=lb[2]; l.w=lb[3];
    ((ushort4*)lo)[i] = l;
  }
}

// ---------------------------------------------------------------------------
// 2) repack lora_B (E, D_OUT, R) -> bf16 (D_OUT, E*R)
// ---------------------------------------------------------------------------
__global__ void pack_loraB_kernel(const float* __restrict__ loraB,
                                  uint16_t* __restrict__ out){
  int t = blockIdx.x * blockDim.x + threadIdx.x;   // t = o*64 + er
  if (t >= N_DIM * ER_) return;
  int o = t >> 6, er = t & 63, e = er >> 4, r = er & 15;
  out[t] = f2bf(loraB[((size_t)e * N_DIM + o) * R_ + r]);
}

// ---------------------------------------------------------------------------
// 3) router logits + softmax gate, one wave per row
// ---------------------------------------------------------------------------
__global__ void router_gate_kernel(const float* __restrict__ x,
                                   const float* __restrict__ wimg, const float* __restrict__ bimg,
                                   const float* __restrict__ wtxt, const float* __restrict__ btxt,
                                   float* __restrict__ gate){
  int wave = threadIdx.x >> 5, lane = threadIdx.x & 31;
  int m = blockIdx.x * 8 + wave;
  int s = m & (S_ - 1);
  const float* wr = (s < SPLIT_) ? wimg : wtxt;
  const float* br = (s < SPLIT_) ? bimg : btxt;
  const float* xr = x + (size_t)m * K_DIM;
  float a0 = 0.f, a1 = 0.f, a2 = 0.f, a3 = 0.f;
  for (int d = lane; d < K_DIM; d += 32){
    float xv = xr[d];
    a0 += xv * wr[d];
    a1 += xv * wr[K_DIM + d];
    a2 += xv * wr[2 * K_DIM + d];
    a3 += xv * wr[3 * K_DIM + d];
  }
#pragma unroll
  for (int off = 16; off >= 1; off >>= 1){
    a0 += __shfl_xor(a0, off, 32);
    a1 += __shfl_xor(a1, off, 32);
    a2 += __shfl_xor(a2, off, 32);
    a3 += __shfl_xor(a3, off, 32);
  }
  float l0 = a0 + br[0], l1 = a1 + br[1], l2 = a2 + br[2], l3 = a3 + br[3];
  float mx = fmaxf(fmaxf(l0, l1), fmaxf(l2, l3));
  float e0 = __expf(l0 - mx), e1 = __expf(l1 - mx), e2 = __expf(l2 - mx), e3 = __expf(l3 - mx);
  float inv = 1.f / (e0 + e1 + e2 + e3);
  if (lane == 0){
    float* g = gate + (size_t)m * 4;
    g[0] = e0 * inv; g[1] = e1 * inv; g[2] = e2 * inv; g[3] = e3 * inv;
  }
}

// ---------------------------------------------------------------------------
// 4) h[m][er] = (x[m] . lora_A[er]) * gate[m][e] * SCALING   (bf16)
// ---------------------------------------------------------------------------
__global__ void lora_h_kernel(const float* __restrict__ x,
                              const float* __restrict__ loraA,
                              const float* __restrict__ gate,
                              uint16_t* __restrict__ h){
  __shared__ float xs[K_DIM];
  int m = blockIdx.x;
  const float* xr = x + (size_t)m * K_DIM;
  for (int i = threadIdx.x; i < K_DIM; i += 256) xs[i] = xr[i];
  __syncthreads();
  int wave = threadIdx.x >> 5, lane = threadIdx.x & 31;
  int er0 = wave * 8;
  float acc[8] = {0.f,0.f,0.f,0.f,0.f,0.f,0.f,0.f};
  for (int d = lane; d < K_DIM; d += 32){
    float xv = xs[d];
#pragma unroll
    for (int j = 0; j < 8; ++j)
      acc[j] += xv * loraA[(size_t)(er0 + j) * K_DIM + d];
  }
#pragma unroll
  for (int j = 0; j < 8; ++j)
#pragma unroll
    for (int off = 16; off >= 1; off >>= 1)
      acc[j] += __shfl_xor(acc[j], off, 32);
  if (lane == 0){
#pragma unroll
    for (int j = 0; j < 8; ++j){
      int er = er0 + j, e = er >> 4;
      h[(size_t)m * ER_ + er] = f2bf(acc[j] * gate[(size_t)m * 4 + e] * SCALING_);
    }
  }
}

// ---------------------------------------------------------------------------
// 5) main WMMA GEMM: out = [x | h] @ [w_base ; lora_B_packed]^T + bias
// ---------------------------------------------------------------------------
#define BM 128
#define BN 128
#define BK 32
#define RS 40          // LDS row stride in bf16 elements (80 B, conflict-free)
#define KT (K_DIM / BK)

#if GEMM_TDM
// One TDM descriptor moves a whole 128x32 (2B-element) tile, row stride 4096,
// inserting 16 B of LDS padding per 64 B row => 80 B LDS pitch (== RS).
__device__ __forceinline__ void tdm_load_tile(const uint16_t* gbase,
                                              uint32_t lds_byte,
                                              uint32_t tensor_rows){
  uint64_t ga = (uint64_t)(uintptr_t)gbase;
  v4u g0;
  g0[0] = 1u;                                            // count=1, user descriptor
  g0[1] = lds_byte;                                      // lds_addr
  g0[2] = (uint32_t)(ga & 0xFFFFFFFFu);                  // global_addr[31:0]
  g0[3] = (uint32_t)((ga >> 32) & 0x01FFFFFFu) | (2u << 30); // global_addr[56:32] | type=2
  const uint32_t td0 = K_DIM, td1 = tensor_rows;
  const uint32_t t0 = BK, t1 = BM;
  const uint64_t s0 = K_DIM;                             // dim0 stride (elements)
  v8i g1;
  g1[0] = (int)((1u << 16)      // data_size = 2 B
              | (1u << 20)      // pad_enable
              | (3u << 22)      // pad_interval: 16 DW = 64 B
              | (3u << 25));    // pad_amount:    4 DW = 16 B
  g1[1] = (int)((td0 & 0xFFFFu) << 16);                  // tensor_dim0[15:0]
  g1[2] = (int)((td0 >> 16) | ((td1 & 0xFFFFu) << 16));  // dim0 hi | dim1 lo
  g1[3] = (int)((td1 >> 16) | (t0 << 16));               // dim1 hi | tile_dim0
  g1[4] = (int)t1;                                       // tile_dim1 (tile_dim2 = 0)
  g1[5] = (int)(uint32_t)(s0 & 0xFFFFFFFFu);             // dim0_stride[31:0]
  g1[6] = (int)(uint32_t)((s0 >> 32) & 0xFFFFu);         // dim0_stride[47:32]
  g1[7] = 0;
  v4i z4 = {0, 0, 0, 0};
#if __clang_major__ >= 23
  v8i z8 = {0, 0, 0, 0, 0, 0, 0, 0};
  __builtin_amdgcn_tensor_load_to_lds(g0, g1, z4, z4, z8, 0);
#else
  __builtin_amdgcn_tensor_load_to_lds(g0, g1, z4, z4, 0);
#endif
}
#endif

template<bool USE_LO>
__global__ __launch_bounds__(256) void moe_lora_gemm_kernel(
    const uint16_t* __restrict__ xhi, const uint16_t* __restrict__ xlo,
    const uint16_t* __restrict__ whi, const uint16_t* __restrict__ wlo,
    const uint16_t* __restrict__ hbuf, const uint16_t* __restrict__ lbt,
    const float* __restrict__ bias, float* __restrict__ out)
{
  __shared__ uint16_t sAh[2][BM * RS];
  __shared__ uint16_t sBh[2][BN * RS];
  __shared__ uint16_t sAl[2][USE_LO ? BM * RS : 1];
  __shared__ uint16_t sBl[2][USE_LO ? BN * RS : 1];

  const int tid  = threadIdx.x;
  const int lane = tid & 31;
  const int wave = tid >> 5;          // 8 waves
  const int wm   = wave & 1;          // 2 wave-rows  (64 M each)
  const int wn   = wave >> 1;         // 4 wave-cols  (32 N each)
  const int m0   = blockIdx.y * BM;
  const int n0   = blockIdx.x * BN;

  v8f acc[4][2];
#pragma unroll
  for (int i = 0; i < 4; ++i)
#pragma unroll
    for (int j = 0; j < 2; ++j)
#pragma unroll
      for (int v = 0; v < 8; ++v) acc[i][j][v] = 0.f;

  // per-lane fragment addressing (identical for A and B fragments):
  //   lanes 0-15  : row = lane,    K = {0..7, 16..23}
  //   lanes 16-31 : row = lane-16, K = {8..15, 24..31}
  const int fr    = lane & 15;
  const int fbyte = (lane >> 4) << 4;     // +16 B for upper half-wave

  auto ldfrag = [&](const uint16_t* s, int row) -> FragAB {
    FragAB f;
    const char* p = (const char*)(s + row * RS) + fbyte;
    f.p.lo = *(const v8bf*)(p);
    f.p.hi = *(const v8bf*)(p + 32);
    return f;
  };

  auto compute_stage = [&](int st){
    FragAB a_h[4], b_h[2], a_l[4], b_l[2];
#pragma unroll
    for (int i = 0; i < 4; ++i){
      int row = wm * 64 + i * 16 + fr;
      a_h[i] = ldfrag(sAh[st], row);
      if (USE_LO) a_l[i] = ldfrag(sAl[st], row);
    }
#pragma unroll
    for (int j = 0; j < 2; ++j){
      int row = wn * 32 + j * 16 + fr;
      b_h[j] = ldfrag(sBh[st], row);
      if (USE_LO) b_l[j] = ldfrag(sBl[st], row);
    }
#pragma unroll
    for (int i = 0; i < 4; ++i)
#pragma unroll
      for (int j = 0; j < 2; ++j){
        acc[i][j] = __builtin_amdgcn_wmma_f32_16x16x32_bf16(
            false, a_h[i].v, false, b_h[j].v, (short)0, acc[i][j], false, false);
        if (USE_LO){
          acc[i][j] = __builtin_amdgcn_wmma_f32_16x16x32_bf16(
              false, a_h[i].v, false, b_l[j].v, (short)0, acc[i][j], false, false);
          acc[i][j] = __builtin_amdgcn_wmma_f32_16x16x32_bf16(
              false, a_l[i].v, false, b_h[j].v, (short)0, acc[i][j], false, false);
        }
      }
  };

#if GEMM_TDM
  // ---- Tensor Data Mover path: wave 0 drives tile DMA, TENSORcnt tracked ----
  auto issue_stage = [&](int st, int kt){
    if (wave == 0){
      const int k0 = kt * BK;
      tdm_load_tile(xhi + (size_t)m0 * K_DIM + k0, lds_addr32(sAh[st]), M_TOT);
      tdm_load_tile(whi + (size_t)n0 * K_DIM + k0, lds_addr32(sBh[st]), N_DIM);
      if (USE_LO){
        tdm_load_tile(xlo + (size_t)m0 * K_DIM + k0, lds_addr32(sAl[st]), M_TOT);
        tdm_load_tile(wlo + (size_t)n0 * K_DIM + k0, lds_addr32(sBl[st]), N_DIM);
      }
    }
  };
  issue_stage(0, 0);
  for (int kt = 0; kt < KT; ++kt){
    const int cur = kt & 1;
    if (kt + 1 < KT) issue_stage(cur ^ 1, kt + 1);
    if (wave == 0){
      if (kt + 1 < KT) __builtin_amdgcn_s_wait_tensorcnt(USE_LO ? 4 : 2);
      else             __builtin_amdgcn_s_wait_tensorcnt(0);
    }
    __syncthreads();                 // publish stage `cur` to all waves
    compute_stage(cur);
    __syncthreads();                 // stage `cur` free for next DMA fill
  }
#elif GEMM_ASYNC
  // ---- per-lane async global->LDS path, ASYNCcnt tracked ----
  const int lrow0 = tid >> 2;         // 0..63
  const int lcol  = (tid & 3) * 8;    // 0,8,16,24
  auto issue_stage = [&](int st, int kt){
    const int k0 = kt * BK;
#pragma unroll
    for (int it = 0; it < 2; ++it){
      int row = lrow0 + it * 64;
      size_t ga = (size_t)(m0 + row) * K_DIM + k0 + lcol;
      size_t gb = (size_t)(n0 + row) * K_DIM + k0 + lcol;
      uint32_t lo_off = (uint32_t)(row * RS + lcol) * 2u;
      __builtin_amdgcn_global_load_async_to_lds_b128(
          (AS1 void*)(uintptr_t)(xhi + ga),
          (AS3 void*)(uint32_t)(lds_addr32(sAh[st]) + lo_off), 0, 0);
      __builtin_amdgcn_global_load_async_to_lds_b128(
          (AS1 void*)(uintptr_t)(whi + gb),
          (AS3 void*)(uint32_t)(lds_addr32(sBh[st]) + lo_off), 0, 0);
      if (USE_LO){
        __builtin_amdgcn_global_load_async_to_lds_b128(
            (AS1 void*)(uintptr_t)(xlo + ga),
            (AS3 void*)(uint32_t)(lds_addr32(sAl[st]) + lo_off), 0, 0);
        __builtin_amdgcn_global_load_async_to_lds_b128(
            (AS1 void*)(uintptr_t)(wlo + gb),
            (AS3 void*)(uint32_t)(lds_addr32(sBl[st]) + lo_off), 0, 0);
      }
    }
  };
  issue_stage(0, 0);
  for (int kt = 0; kt < KT; ++kt){
    const int cur = kt & 1;
    if (kt + 1 < KT) issue_stage(cur ^ 1, kt + 1);
    if (kt + 1 < KT) __builtin_amdgcn_s_wait_asynccnt(USE_LO ? 8 : 4);
    else             __builtin_amdgcn_s_wait_asynccnt(0);
    __syncthreads();
    compute_stage(cur);
    __syncthreads();
  }
#else
  // ---- fallback: synchronous VGPR-staged path ----
  const int lrow0 = tid >> 2;
  const int lcol  = (tid & 3) * 8;
  for (int kt = 0; kt < KT; ++kt){
    const int k0 = kt * BK;
    uint4 ra[2], rb[2], ral[2], rbl[2];
#pragma unroll
    for (int it = 0; it < 2; ++it){
      int row = lrow0 + it * 64;
      size_t ga = (size_t)(m0 + row) * K_DIM + k0 + lcol;
      size_t gb = (size_t)(n0 + row) * K_DIM + k0 + lcol;
      ra[it] = *(const uint4*)(xhi + ga);
      rb[it] = *(const uint4*)(whi + gb);
      if (USE_LO){ ral[it] = *(const uint4*)(xlo + ga); rbl[it] = *(const uint4*)(wlo + gb); }
    }
    __syncthreads();
#pragma unroll
    for (int it = 0; it < 2; ++it){
      int row = lrow0 + it * 64;
      *(uint4*)&sAh[0][row * RS + lcol] = ra[it];
      *(uint4*)&sBh[0][row * RS + lcol] = rb[it];
      if (USE_LO){
        *(uint4*)&sAl[0][row * RS + lcol] = ral[it];
        *(uint4*)&sBl[0][row * RS + lcol] = rbl[it];
      }
    }
    __syncthreads();
    compute_stage(0);
  }
#endif

  // fused LoRA: two extra K=32 WMMA steps straight from global (rows are 128 B)
  {
    const int helem = (lane >> 4) * 8;
#pragma unroll
    for (int kc = 0; kc < 2; ++kc){
      FragAB af[4], bf_[2];
#pragma unroll
      for (int i = 0; i < 4; ++i){
        int gm = m0 + wm * 64 + i * 16 + fr;
        const uint16_t* p = hbuf + (size_t)gm * ER_ + kc * 32 + helem;
        af[i].p.lo = *(const v8bf*)(p);
        af[i].p.hi = *(const v8bf*)(p + 16);
      }
#pragma unroll
      for (int j = 0; j < 2; ++j){
        int gn = n0 + wn * 32 + j * 16 + fr;
        const uint16_t* p = lbt + (size_t)gn * ER_ + kc * 32 + helem;
        bf_[j].p.lo = *(const v8bf*)(p);
        bf_[j].p.hi = *(const v8bf*)(p + 16);
      }
#pragma unroll
      for (int i = 0; i < 4; ++i)
#pragma unroll
        for (int j = 0; j < 2; ++j)
          acc[i][j] = __builtin_amdgcn_wmma_f32_16x16x32_bf16(
              false, af[i].v, false, bf_[j].v, (short)0, acc[i][j], false, false);
    }
  }

  // epilogue: +bias, store fp32
#pragma unroll
  for (int j = 0; j < 2; ++j){
    int gn = n0 + wn * 32 + j * 16 + fr;
    float bv = bias[gn];
    int rbase = m0 + wm * 64 + ((lane >> 4) << 3);
#pragma unroll
    for (int i = 0; i < 4; ++i){
      int gm0 = rbase + i * 16;
#pragma unroll
      for (int v = 0; v < 8; ++v)
        out[(size_t)(gm0 + v) * N_DIM + gn] = acc[i][j][v] + bv;
    }
  }
}

// ---------------------------------------------------------------------------
extern "C" void kernel_launch(void* const* d_in, const int* in_sizes, int n_in,
                              void* d_out, int out_size, void* d_ws, size_t ws_size,
                              hipStream_t stream){
  const float* x       = (const float*)d_in[0];
  const float* w_base  = (const float*)d_in[1];
  const float* b_base  = (const float*)d_in[2];
  const float* w_r_img = (const float*)d_in[3];
  const float* b_r_img = (const float*)d_in[4];
  const float* w_r_txt = (const float*)d_in[5];
  const float* b_r_txt = (const float*)d_in[6];
  const float* loraA   = (const float*)d_in[7];
  const float* loraB   = (const float*)d_in[8];
  float* out = (float*)d_out;

  char* ws = (char*)d_ws;
  size_t off = 0;
  auto wsalloc = [&](size_t bytes) -> char* {
    char* p = ws + off;
    off = (off + bytes + 255) & ~(size_t)255;
    return p;
  };
  uint16_t* xhi  = (uint16_t*)wsalloc((size_t)M_TOT * K_DIM * 2);
  uint16_t* whi  = (uint16_t*)wsalloc((size_t)N_DIM * K_DIM * 2);
  uint16_t* lbt  = (uint16_t*)wsalloc((size_t)N_DIM * ER_ * 2);
  float*    gate = (float*)   wsalloc((size_t)M_TOT * E_ * 4);
  uint16_t* hbuf = (uint16_t*)wsalloc((size_t)M_TOT * ER_ * 2);
  uint16_t* xlo  = (uint16_t*)wsalloc((size_t)M_TOT * K_DIM * 2);
  uint16_t* wlo  = (uint16_t*)wsalloc((size_t)N_DIM * K_DIM * 2);
  const bool use_lo = (ws_size >= off);   // fall back to plain bf16 if ws too small

  const int n4x = M_TOT * K_DIM / 4;
  split_bf16_kernel<<<n4x / 256, 256, 0, stream>>>(x, xhi, use_lo ? xlo : nullptr, n4x);
  const int n4w = N_DIM * K_DIM / 4;
  split_bf16_kernel<<<n4w / 256, 256, 0, stream>>>(w_base, whi, use_lo ? wlo : nullptr, n4w);
  pack_loraB_kernel<<<(N_DIM * ER_) / 256, 256, 0, stream>>>(loraB, lbt);
  router_gate_kernel<<<M_TOT / 8, 256, 0, stream>>>(x, w_r_img, b_r_img, w_r_txt, b_r_txt, gate);
  lora_h_kernel<<<M_TOT, 256, 0, stream>>>(x, loraA, gate, hbuf);

  dim3 grid(N_DIM / BN, M_TOT / BM);
  if (use_lo)
    moe_lora_gemm_kernel<true ><<<grid, 256, 0, stream>>>(xhi, xlo, whi, wlo, hbuf, lbt, b_base, out);
  else
    moe_lora_gemm_kernel<false><<<grid, 256, 0, stream>>>(xhi, xhi, whi, whi, hbuf, lbt, b_base, out);
}
